// MultiHeadAttention_74440373174607
// MI455X (gfx1250) — matrix operand order repaired and checked by reading, not compiled, not run on hardware
//
#include <hip/hip_runtime.h>
#include <hip/hip_bf16.h>

// ---------------------------------------------------------------------------
// MHA for MI455X (gfx1250): f16 WMMA everywhere, flash-attention streaming.
// Round 3: K/V chunks staged in LDS once per block (4 waves share one (b,h)),
// double-buffered with one barrier per 32-key chunk -> L2 traffic / 4.
// ---------------------------------------------------------------------------

typedef _Float16 half_t;
typedef __attribute__((ext_vector_type(2)))  _Float16 v2h;
typedef __attribute__((ext_vector_type(8)))  _Float16 v8h;
typedef __attribute__((ext_vector_type(16))) _Float16 v16h;
typedef __attribute__((ext_vector_type(8)))  float    v8f;

#define DIM    768
#define SEQ    4096
#define BATCH  2
#define HEADS  12
#define HD     64
#define ROWS   (BATCH * SEQ)          // 8192 tokens
#define SCALE  0.125f                 // 64^-0.5

#define KSTRIDE 72                    // LDS K row stride (halfs): 36 dwords
#define VSTRIDE 40                    // LDS V row stride (halfs): 20 dwords

static __device__ __forceinline__ v8f wmma_f16(v16h a, v16h b, v8f c) {
  return __builtin_amdgcn_wmma_f32_16x16x32_f16(
      false, a, false, b, (short)0, c, false, false);
}

// XOR-butterfly across lanes via ds_swizzle (group-of-32: and=0x1f, xor=MASK).
template <int MASK>
static __device__ __forceinline__ float bfly(float x) {
  int i = __builtin_amdgcn_ds_swizzle(__builtin_bit_cast(int, x),
                                      0x001f | (MASK << 10));
  return __builtin_bit_cast(float, i);
}

// 16-byte-granular v16h copy helpers (no 32B alignment assumption).
static __device__ __forceinline__ v16h ld_v16h_u16(const half_t* p) {
  v8h lo = *(const v8h*)p;
  v8h hi = *(const v8h*)(p + 8);
  v16h r;
#pragma unroll
  for (int i = 0; i < 8; ++i) { r[i] = lo[i]; r[8 + i] = hi[i]; }
  return r;
}
static __device__ __forceinline__ void st_v16h_u16(half_t* p, v16h x) {
  v8h lo, hi;
#pragma unroll
  for (int i = 0; i < 8; ++i) { lo[i] = x[i]; hi[i] = x[8 + i]; }
  *(v8h*)p = lo;
  *(v8h*)(p + 8) = hi;
}

// ---------------------------------------------------------------------------
// Fragment loaders (wave32; A and B both use contiguous k = 16*halfsel + s, not the ISA table order).
// A 16x32 f16: lane = M; per lane 16 contiguous halfs at
//   k0 + halfsel*16 + {0..15}.
// B 32x16 f16: lane = N; lanes 0-15 hold K=k0..k0+15, lanes 16-31 K=+16..31.
// C/D 16x16 f32: VGPR r -> M = r + 8*(lane>=16), N = lane&15.
// ---------------------------------------------------------------------------
static __device__ __forceinline__ v16h load_frag_a(const half_t* __restrict__ base,
                                                   int row0, int k0, int lda) {
  int lane = threadIdx.x & 31;
  const half_t* p = base + (size_t)(row0 + (lane & 15)) * lda + k0 + ((lane >> 4) << 4);
  v16h a;
#pragma unroll
  for (int g = 0; g < 8; ++g) {
    int k = g << 1;   // {0,2,4,...,14}
    a[2 * g]     = p[k];
    a[2 * g + 1] = p[k + 1];
  }
  return a;
}

// B fragment where element (k,n) lives at src[n*ld + k] (B^T row-major),
// contiguous 32B per lane. Global-memory version (32B aligned).
static __device__ __forceinline__ v16h load_frag_bt(const half_t* __restrict__ base,
                                                    int k0, int col0, int ld) {
  int lane = threadIdx.x & 31;
  const half_t* p = base + (size_t)(col0 + (lane & 15)) * ld + k0 + ((lane >> 4) << 4);
  return *(const v16h*)p;
}
// LDS version (16B granular).
static __device__ __forceinline__ v16h load_frag_bt_lds(const half_t* base,
                                                        int k0, int col0, int ld) {
  int lane = threadIdx.x & 31;
  const half_t* p = base + (size_t)(col0 + (lane & 15)) * ld + k0 + ((lane >> 4) << 4);
  return ld_v16h_u16(p);
}

// ---------------------------------------------------------------------------
// Conversions
// ---------------------------------------------------------------------------
__global__ void cvt_f32_to_f16(const float* __restrict__ in,
                               half_t* __restrict__ out, int n) {
  int i = blockIdx.x * blockDim.x + threadIdx.x;
  if (i < n) out[i] = (half_t)in[i];
}

// W[k][n] (f32, DIMxDIM) -> WT[n][k] (f16).
__global__ void cvt_transpose_f16(const float* __restrict__ in,
                                  half_t* __restrict__ out) {
  int i = blockIdx.x * blockDim.x + threadIdx.x;   // i = k*DIM + n
  int k = i / DIM, n = i % DIM;
  out[(size_t)n * DIM + k] = (half_t)in[i];
}

// ---------------------------------------------------------------------------
// GEMM: C[M,N] = A[M,K] * W[K,N] + bias (WT passed transposed: [N][K] f16).
// Per-wave 32x32 tile, 8 waves -> 128x64 block tile.
// MODE 0: f16 scatter into [B,H,N,Hd] (Q with scale folded, K).
// MODE 2: f16 scatter into key-permuted transposed V layout [B,H,Hd,N'].
// MODE 1: f32 flat [M,N] (output projection -> d_out).
// ---------------------------------------------------------------------------
template <int MODE>
__global__ __launch_bounds__(256) void gemm_kernel(
    const half_t* __restrict__ A, const half_t* __restrict__ WT,
    const float* __restrict__ bias, float scale,
    half_t* __restrict__ out16, float* __restrict__ out32,
    int M, int N, int K) {
  int lane = threadIdx.x & 31;
  int wave = threadIdx.x >> 5;
  int row0 = blockIdx.x * 128 + (wave >> 1) * 32;
  int col0 = blockIdx.y * 64 + (wave & 1) * 32;

  v8f acc[2][2] = {};
  for (int k0 = 0; k0 < K; k0 += 32) {
    v16h a0 = load_frag_a(A, row0, k0, K);
    v16h a1 = load_frag_a(A, row0 + 16, k0, K);
    v16h b0 = load_frag_bt(WT, k0, col0, K);
    v16h b1 = load_frag_bt(WT, k0, col0 + 16, K);
    acc[0][0] = wmma_f16(a0, b0, acc[0][0]);
    acc[0][1] = wmma_f16(a0, b1, acc[0][1]);
    acc[1][0] = wmma_f16(a1, b0, acc[1][0]);
    acc[1][1] = wmma_f16(a1, b1, acc[1][1]);
  }

  int halfsel = lane >> 4;
#pragma unroll
  for (int mi = 0; mi < 2; ++mi)
#pragma unroll
    for (int ni = 0; ni < 2; ++ni)
#pragma unroll
      for (int r = 0; r < 8; ++r) {
        int m = row0 + mi * 16 + r + 8 * halfsel;
        int c = col0 + ni * 16 + (lane & 15);
        float val = (acc[mi][ni][r] + bias[c]) * scale;
        if (MODE == 0) {
          int b = m >> 12, n = m & (SEQ - 1);
          int h = c >> 6, hd = c & (HD - 1);
          out16[(((size_t)(b * HEADS + h) * SEQ + n) << 6) + hd] = (half_t)val;
        } else if (MODE == 2) {
          // V layout [B,H,Hd,N'], key order permuted per 32: pi(j)=(j&1)*16+(j>>1)
          int b = m >> 12, n = m & (SEQ - 1);
          int h = c >> 6, hd = c & (HD - 1);
          int j = (((n & 15) << 1) | ((n >> 4) & 1));
          out16[((size_t)(b * HEADS + h) * HD + hd) * SEQ + (n & ~31) + j] =
              (half_t)val;
        } else {
          out32[(size_t)m * N + c] = val;
        }
      }
}

// ---------------------------------------------------------------------------
// Flash attention. Block = 4 waves, all sharing one (b,h); each wave owns a
// 16-query tile. 32-key K/V chunks are staged in LDS once per block
// (double-buffered, coalesced cooperative loads, one barrier per chunk).
// Online softmax with ds_swizzle butterflies; P goes D->A layout through a
// per-wave LDS tile (packed b32 stores thanks to the key permutation).
// ---------------------------------------------------------------------------
__global__ __launch_bounds__(128) void attn_kernel(
    const half_t* __restrict__ q, const half_t* __restrict__ k,
    const half_t* __restrict__ v, half_t* __restrict__ out) {
  __shared__ half_t kbuf[2][32][KSTRIDE];   // 32 keys x 64 d (+pad)
  __shared__ half_t vbuf[2][64][VSTRIDE];   // 64 d x 32 keys' (+pad)
  __shared__ half_t pbuf[4][16][40];        // per-wave 16x32 P tile (+pad)

  int tid = threadIdx.x;
  int lane = tid & 31;
  int wave = tid >> 5;
  int qt = blockIdx.x * 4 + wave;      // all 4 waves share qt>>8
  int bh = qt >> 8;                    // (b*HEADS + h)
  int qrow0 = (qt & 255) << 4;

  const half_t* qb = q + ((size_t)bh * SEQ << 6);   // [N, Hd]
  const half_t* kb = k + ((size_t)bh * SEQ << 6);   // [N, Hd]
  const half_t* vb = v + ((size_t)bh * SEQ << 6);   // [Hd, N'] permuted

  // cooperative staging assignment: 2048 halfs per chunk, 16 halfs per thread
  int krow = tid >> 2, kcol = (tid & 3) << 4;       // K chunk: [32][64]
  int vrow = tid >> 1, vcol = (tid & 1) << 4;       // V chunk: [64][32]

  // Q tile (pre-scaled by SCALE at projection): 16x64 in registers.
  v16h qa0 = load_frag_a(qb, qrow0, 0, HD);
  v16h qa1 = load_frag_a(qb, qrow0, 32, HD);

  // preload chunk 0
  {
    v16h kx = *(const v16h*)(kb + (size_t)krow * HD + kcol);
    v16h vx = *(const v16h*)(vb + (size_t)vrow * SEQ + vcol);
    st_v16h_u16(&kbuf[0][krow][kcol], kx);
    st_v16h_u16(&vbuf[0][vrow][vcol], vx);
  }
  __syncthreads();

  v8f o[4] = {};
  float mrow[8], lrow[8];
#pragma unroll
  for (int r = 0; r < 8; ++r) { mrow[r] = -1e30f; lrow[r] = 0.f; }

  int row_off = (lane >> 4) << 3;      // D-layout: row = r + 8*halfsel

  for (int it = 0; it < SEQ / 32; ++it) {
    int cur = it & 1, nxt = cur ^ 1;
    int key0 = it << 5;
    bool has_next = (it + 1) < (SEQ / 32);

    // prefetch next chunk into registers (uniform branch)
    v16h kx{}, vx{};
    if (has_next) {
      kx = *(const v16h*)(kb + (size_t)(key0 + 32 + krow) * HD + kcol);
      vx = *(const v16h*)(vb + (size_t)vrow * SEQ + key0 + 32 + vcol);
    }

    const half_t* kc = &kbuf[cur][0][0];
    const half_t* vc = &vbuf[cur][0][0];

    // ---- S = Q @ K^T for two 16-key tiles (from LDS) ---------------------
    v8f s0 = {}, s1 = {};
    {
      v16h kb0 = load_frag_bt_lds(kc, 0, 0, KSTRIDE);
      v16h kb1 = load_frag_bt_lds(kc, 32, 0, KSTRIDE);
      s0 = wmma_f16(qa0, kb0, s0);
      s0 = wmma_f16(qa1, kb1, s0);
      v16h kb2 = load_frag_bt_lds(kc, 0, 16, KSTRIDE);
      v16h kb3 = load_frag_bt_lds(kc, 32, 16, KSTRIDE);
      s1 = wmma_f16(qa0, kb2, s1);
      s1 = wmma_f16(qa1, kb3, s1);
    }

    // ---- online softmax over the 32-key chunk ----------------------------
#pragma unroll
    for (int r = 0; r < 8; ++r) {
      float a = s0[r];
      float b = s1[r];
      float cm = fmaxf(a, b);
      cm = fmaxf(cm, bfly<1>(cm));
      cm = fmaxf(cm, bfly<2>(cm));
      cm = fmaxf(cm, bfly<4>(cm));
      cm = fmaxf(cm, bfly<8>(cm));
      float newm = fmaxf(mrow[r], cm);
      float p0 = __expf(a - newm);
      float p1 = __expf(b - newm);
      float corr = __expf(mrow[r] - newm);
      float rs = p0 + p1;
      rs += bfly<1>(rs);
      rs += bfly<2>(rs);
      rs += bfly<4>(rs);
      rs += bfly<8>(rs);
      lrow[r] = lrow[r] * corr + rs;
      mrow[r] = newm;
      o[0][r] *= corr; o[1][r] *= corr; o[2][r] *= corr; o[3][r] *= corr;
      // LDS col 2t <-> key key0+t, col 2t+1 <-> key key0+16+t (matches pi)
      int row = r + row_off;
      *(v2h*)&pbuf[wave][row][(lane & 15) << 1] = (v2h){(half_t)p0, (half_t)p1};
    }
    asm volatile("s_wait_dscnt 0x0" ::: "memory");   // intra-wave LDS order

    // ---- O += P @ V (V rows permuted identically to P columns) -----------
    v16h pa = load_frag_a(&pbuf[wave][0][0], 0, 0, 40);
    v16h vb0 = load_frag_bt_lds(vc, 0, 0, VSTRIDE);
    v16h vb1 = load_frag_bt_lds(vc, 0, 16, VSTRIDE);
    v16h vb2 = load_frag_bt_lds(vc, 0, 32, VSTRIDE);
    v16h vb3 = load_frag_bt_lds(vc, 0, 48, VSTRIDE);
    o[0] = wmma_f16(pa, vb0, o[0]);
    o[1] = wmma_f16(pa, vb1, o[1]);
    o[2] = wmma_f16(pa, vb2, o[2]);
    o[3] = wmma_f16(pa, vb3, o[3]);

    // ---- publish next chunk, one barrier per iteration -------------------
    if (has_next) {
      st_v16h_u16(&kbuf[nxt][krow][kcol], kx);
      st_v16h_u16(&vbuf[nxt][vrow][vcol], vx);
    }
    __syncthreads();
  }

  // ---- epilogue: normalize, write [b*SEQ+m, h*64+d] as f16 ----------------
  int b = bh / HEADS, h = bh % HEADS;
#pragma unroll
  for (int ni = 0; ni < 4; ++ni)
#pragma unroll
    for (int r = 0; r < 8; ++r) {
      int m = qrow0 + r + row_off;
      int d = ni * 16 + (lane & 15);
      float val = o[ni][r] / lrow[r];
      out[(size_t)(b * SEQ + m) * DIM + h * HD + d] = (half_t)val;
    }
}

// ---------------------------------------------------------------------------
// Host launcher
// ---------------------------------------------------------------------------
extern "C" void kernel_launch(void* const* d_in, const int* in_sizes, int n_in,
                              void* d_out, int out_size, void* d_ws, size_t ws_size,
                              hipStream_t stream) {
  const float* x  = (const float*)d_in[0];
  const float* Wq = (const float*)d_in[1];
  const float* bq = (const float*)d_in[2];
  const float* Wk = (const float*)d_in[3];
  const float* bk = (const float*)d_in[4];
  const float* Wv = (const float*)d_in[5];
  const float* bv = (const float*)d_in[6];
  const float* Wo = (const float*)d_in[7];
  const float* bo = (const float*)d_in[8];
  float* out = (float*)d_out;

  const size_t XE = (size_t)ROWS * DIM;   // 8192*768
  const size_t WE = (size_t)DIM * DIM;

  char* ws = (char*)d_ws;
  half_t* x16   = (half_t*)ws;            ws += XE * sizeof(half_t);
  half_t* wq16  = (half_t*)ws;            ws += WE * sizeof(half_t);
  half_t* wk16  = (half_t*)ws;            ws += WE * sizeof(half_t);
  half_t* wv16  = (half_t*)ws;            ws += WE * sizeof(half_t);
  half_t* wo16  = (half_t*)ws;            ws += WE * sizeof(half_t);
  half_t* q16   = (half_t*)ws;            ws += XE * sizeof(half_t);
  half_t* k16   = (half_t*)ws;            ws += XE * sizeof(half_t);
  half_t* v16   = (half_t*)ws;            ws += XE * sizeof(half_t);
  half_t* at16  = (half_t*)ws;            ws += XE * sizeof(half_t);

  // 1) convert x to f16; convert+transpose the four weights
  cvt_f32_to_f16<<<(int)((XE + 255) / 256), 256, 0, stream>>>(x, x16, (int)XE);
  cvt_transpose_f16<<<(int)(WE / 256), 256, 0, stream>>>(Wq, wq16);
  cvt_transpose_f16<<<(int)(WE / 256), 256, 0, stream>>>(Wk, wk16);
  cvt_transpose_f16<<<(int)(WE / 256), 256, 0, stream>>>(Wv, wv16);
  cvt_transpose_f16<<<(int)(WE / 256), 256, 0, stream>>>(Wo, wo16);

  // 2) QKV projections (f16 WMMA, f32 accumulate)
  dim3 ggrid(ROWS / 128, DIM / 64);
  gemm_kernel<0><<<ggrid, 256, 0, stream>>>(x16, wq16, bq, SCALE, q16, nullptr,
                                            ROWS, DIM, DIM);
  gemm_kernel<0><<<ggrid, 256, 0, stream>>>(x16, wk16, bk, 1.0f, k16, nullptr,
                                            ROWS, DIM, DIM);
  gemm_kernel<2><<<ggrid, 256, 0, stream>>>(x16, wv16, bv, 1.0f, v16, nullptr,
                                            ROWS, DIM, DIM);

  // 3) flash attention: 6144 query tiles, 4 waves/block share K/V via LDS
  attn_kernel<<<(BATCH * HEADS * (SEQ / 16)) / 4, 128, 0, stream>>>(q16, k16,
                                                                    v16, at16);

  // 4) output projection -> f32 d_out
  gemm_kernel<1><<<ggrid, 256, 0, stream>>>(at16, wo16, bo, 1.0f, nullptr, out,
                                            ROWS, DIM, DIM);
}
